// SSModel_20667382628997
// MI455X (gfx1250) — matrix operand order repaired and checked by reading, not compiled, not run on hardware
//
#include <hip/hip_runtime.h>
#include <hip/hip_bf16.h>
#include <stdint.h>

typedef __bf16 bf16_t;
typedef __attribute__((ext_vector_type(16))) __bf16 v16bf;
typedef __attribute__((ext_vector_type(8)))  __bf16 v8bf;
typedef __attribute__((ext_vector_type(8)))  float  v8f;

#define N_NODES  100000
#define N_EDGES  200000
#define N_GRAPHS 2048
#define FN 39
#define FE 13
#define FG 16
#define OUTC 500

// -------------------- elementwise / scatter kernels --------------------

__global__ void k_f32_to_bf16(const float* __restrict__ in, bf16_t* __restrict__ out,
                              long long n) {
  long long i = (long long)blockIdx.x * blockDim.x + threadIdx.x;
  long long st = (long long)gridDim.x * blockDim.x;
  for (; i < n; i += st) out[i] = (bf16_t)in[i];
}

__global__ void k_zero_f32(float* __restrict__ p, long long n) {
  long long i = (long long)blockIdx.x * blockDim.x + threadIdx.x;
  long long st = (long long)gridDim.x * blockDim.x;
  for (; i < n; i += st) p[i] = 0.0f;
}

// scatter-add bf16 features (rows x 2^lcol) into fp32 accumulator at idx[row], plus counts
__global__ void k_scatter_add(const bf16_t* __restrict__ feat, const int* __restrict__ idx,
                              float* __restrict__ acc, float* __restrict__ cnt,
                              int rows, int lcol) {
  const int ncol = 1 << lcol;
  long long total = (long long)rows << lcol;
  long long t = (long long)blockIdx.x * blockDim.x + threadIdx.x;
  long long st = (long long)gridDim.x * blockDim.x;
  for (; t < total; t += st) {
    int r = (int)(t >> lcol);
    int j = (int)(t & (ncol - 1));
    int d = idx[r];
    atomicAdd(&acc[((long long)d << lcol) + j], (float)feat[t]);
    if (j == 0) atomicAdd(&cnt[d], 1.0f);
  }
}

__global__ void k_mean_to_bf16(const float* __restrict__ acc, const float* __restrict__ cnt,
                               bf16_t* __restrict__ out, int rows, int lcol) {
  long long total = (long long)rows << lcol;
  long long t = (long long)blockIdx.x * blockDim.x + threadIdx.x;
  long long st = (long long)gridDim.x * blockDim.x;
  for (; t < total; t += st) {
    int r = (int)(t >> lcol);
    out[t] = (bf16_t)(acc[t] / fmaxf(cnt[r], 1.0f));
  }
}

// -------------------- fused gather/concat WMMA GEMM --------------------
// Logical A[i,k] = k<s1 ? x0[idx0(i)*ld0+k] : (k<s2 ? x1[idx1(i)*ld1+k-s1]
//                                                   : x2[i*ld2+k-s2])
struct ConcatA {
  const bf16_t* x0; const int* i0; int ld0; int s1;
  const bf16_t* x1; const int* i1; int ld1; int s2;
  const bf16_t* x2; int ld2;
};

// C[M,N] = act(A[M,K] @ W[K,N] + bias), bf16 inputs, fp32 accumulate
// (v_wmma_f32_16x16x32_bf16). Block tile 128x128, 8 waves (wave32),
// wave tile 32x64 = 2x4 subtiles of 16x16. Double-buffered LDS.
template <int RELU, int OUT_BF16>
__global__ __launch_bounds__(256) void k_gemm_wmma(ConcatA A, const bf16_t* __restrict__ W,
                                                   const float* __restrict__ bias,
                                                   void* __restrict__ Cout,
                                                   int M, int N, int K) {
  __shared__ __align__(16) bf16_t As[2][128][40];   // [buf][m][k], +8 pad
  __shared__ __align__(16) bf16_t Bs[2][128][40];   // [buf][n][k] (transposed), +8 pad

  const int tid = threadIdx.x;
  const int bm = blockIdx.y, bn = blockIdx.x;
  const int wave = tid >> 5, lane = tid & 31;
  const int wm = wave >> 1, wn = wave & 1;   // 4x2 wave grid
  const int lg = lane >> 4, ln = lane & 15;

  // ---- A-fill state (wave-uniform k-half: waves 0-3 low, 4-7 high) ----
  const int ar = tid & 127;                   // row within tile, per thread
  const int ahalf = tid >> 7;                 // 16-wide k half, wave-uniform
  const int agrow = bm * 128 + ar;
  const bool arOK = agrow < M;
  long long r0 = 0, r1 = 0;
  if (arOK) {
    r0 = A.i0 ? (long long)A.i0[agrow] : (long long)agrow;
    r1 = A.i1 ? (long long)A.i1[agrow] : (long long)agrow;
  }
  const bf16_t* p0 = A.x0 + r0 * A.ld0;
  const bf16_t* p1 = A.x1 ? (A.x1 + r1 * A.ld1) : A.x0;               // never used if region empty
  const bf16_t* p2 = A.x2 ? (A.x2 + (long long)agrow * A.ld2) : A.x0; // never used if region empty
  const int s1 = A.s1, s2 = A.s2;
  // uniform per-region 16B-alignment predicates (buffers are 256B-aligned, k0 mult of 16)
  const bool al0 = (A.ld0 & 7) == 0;
  const bool al1 = ((A.ld1 & 7) == 0) && ((s1 & 7) == 0);
  const bool al2 = ((A.ld2 & 7) == 0) && ((s2 & 7) == 0);

  // ---- B-fill state ----
  const int bkk = tid >> 3;                  // 0..31  (k within tile)
  const int bnb = tid & 7;                   // 0..7   (16-wide n chunk)
  const int bgn0 = bn * 128 + bnb * 16;
  const bool bAligned = (N & 7) == 0;        // uniform: W rows 16B-aligned for all gk

  auto fill = [&](int kt, int buf) {
    // ---- A tile: per-thread 16 contiguous k, uniform path select ----
    {
      const int k0 = (kt << 5) + ahalf * 16;   // wave-uniform
      bf16_t* dst = &As[buf][ar][ahalf * 16];
      // uniform region classification for segment [k0, k0+16)
      bool fast = false;
      const bf16_t* src = p0;
      if (k0 + 16 <= s1)                     { fast = al0; src = p0 + k0; }
      else if (k0 >= s1 && k0 + 16 <= s2)    { fast = al1; src = p1 + (k0 - s1); }
      else if (k0 >= s2 && k0 + 16 <= K)     { fast = al2; src = p2 + (k0 - s2); }
      if (fast) {                              // scalar branch, no exec divergence
        v8bf lo = {}, hi = {};
        if (arOK) {                            // exec-masked b128 loads
          lo = *(const v8bf*)src;
          hi = *(const v8bf*)(src + 8);
        }
        *(v8bf*)dst       = lo;
        *(v8bf*)(dst + 8) = hi;
      } else {
        // scalar path: branch-free address select + one guarded load per element
        #pragma unroll
        for (int t = 0; t < 16; ++t) {
          const int gk = k0 + t;
          const bf16_t* q = p2 + (gk - s2);
          q = (gk < s2) ? (p1 + (gk - s1)) : q;
          q = (gk < s1) ? (p0 + gk) : q;
          bf16_t v = (bf16_t)0.0f;
          if (arOK && gk < K) v = *q;
          dst[t] = v;
        }
      }
    }
    // ---- B tile: wide global loads, transposed LDS writes ----
    {
      const int gk = (kt << 5) + bkk;
      const bool kOK = gk < K;
      if (bAligned) {                          // uniform branch
        const bf16_t* wsrc = W + (long long)gk * N + bgn0;
        v8bf a = {}, b = {};
        if (kOK && bgn0 + 16 <= N) {           // exec-masked b128 loads
          a = *(const v8bf*)wsrc;
          b = *(const v8bf*)(wsrc + 8);
        }
        #pragma unroll
        for (int i = 0; i < 8; ++i) {
          Bs[buf][bnb * 16 + i][bkk]     = a[i];
          Bs[buf][bnb * 16 + 8 + i][bkk] = b[i];
        }
      } else {
        const unsigned* wrow = (const unsigned*)(W + (long long)gk * N);  // gk*N always even
        #pragma unroll
        for (int t = 0; t < 8; ++t) {
          const int gn = bgn0 + 2 * t;
          unsigned pk = 0u;
          if (kOK && gn < N) pk = wrow[gn >> 1];
          Bs[buf][bnb * 16 + 2 * t][bkk]     = ((const bf16_t*)&pk)[0];
          Bs[buf][bnb * 16 + 2 * t + 1][bkk] = ((const bf16_t*)&pk)[1];
        }
      }
    }
  };

  v8f acc[2][4] = {};
  const int KT = (K + 31) >> 5;

  fill(0, 0);
  for (int kt = 0; kt < KT; ++kt) {
    __syncthreads();                 // tile kt ready; prior reads of buf^1 done
    const int buf = kt & 1;
    if (kt + 1 < KT) fill(kt + 1, buf ^ 1);

    // ---- fragments (ISA 16-bit A 16x32 / B 32x16 lane layouts), b128 LDS reads ----
    v16bf af[2];
    #pragma unroll
    for (int tm = 0; tm < 2; ++tm) {
      const bf16_t* ap = &As[buf][wm * 32 + tm * 16 + ln][lg * 8];
      v8bf lo = *(const v8bf*)ap;          // K = lg*8 .. lg*8+7
      v8bf hi = *(const v8bf*)(ap + 16);   // K = lg*8+16 .. lg*8+23
      #pragma unroll
      for (int i = 0; i < 8; ++i) { af[tm][i] = lo[i]; af[tm][i + 8] = hi[i]; }
    }
    v16bf bfr[4];
    #pragma unroll
    for (int tn = 0; tn < 4; ++tn) {
      const bf16_t* bp = &Bs[buf][wn * 64 + tn * 16 + ln][lg * 8];
      v8bf lo = *(const v8bf*)bp;          // K = lg*8 .. lg*8+7
      v8bf hi = *(const v8bf*)(bp + 16);   // K = lg*8+16 .. lg*8+23
      #pragma unroll
      for (int i = 0; i < 8; ++i) { bfr[tn][i] = lo[i]; bfr[tn][i + 8] = hi[i]; }
    }

    #pragma unroll
    for (int tm = 0; tm < 2; ++tm)
      #pragma unroll
      for (int tn = 0; tn < 4; ++tn)
        acc[tm][tn] = __builtin_amdgcn_wmma_f32_16x16x32_bf16(
            false, af[tm], false, bfr[tn], (short)0, acc[tm][tn], false, false);
  }

  // ---- epilogue: bias + optional relu, fp32 or bf16 output ----
  #pragma unroll
  for (int tm = 0; tm < 2; ++tm) {
    #pragma unroll
    for (int tn = 0; tn < 4; ++tn) {
      const int col = bn * 128 + wn * 64 + tn * 16 + ln;
      #pragma unroll
      for (int r = 0; r < 8; ++r) {
        const int row = bm * 128 + wm * 32 + tm * 16 + lg * 8 + r;
        if (row < M && col < N) {
          float v = acc[tm][tn][r] + bias[col];
          if (RELU) v = fmaxf(v, 0.0f);
          if (OUT_BF16) ((bf16_t*)Cout)[(long long)row * N + col] = (bf16_t)v;
          else          ((float*)Cout)[(long long)row * N + col] = v;
        }
      }
    }
  }
}

// -------------------- log_softmax, block per row --------------------
__global__ __launch_bounds__(256) void k_log_softmax(const float* __restrict__ in,
                                                     float* __restrict__ out, int C) {
  __shared__ float red[256];
  int row = blockIdx.x, tid = threadIdx.x;
  const float* x = in + (long long)row * C;
  float m = -3.0e38f;
  for (int c = tid; c < C; c += 256) m = fmaxf(m, x[c]);
  red[tid] = m; __syncthreads();
  for (int s = 128; s > 0; s >>= 1) { if (tid < s) red[tid] = fmaxf(red[tid], red[tid + s]); __syncthreads(); }
  m = red[0]; __syncthreads();
  float sum = 0.0f;
  for (int c = tid; c < C; c += 256) sum += __expf(x[c] - m);
  red[tid] = sum; __syncthreads();
  for (int s = 128; s > 0; s >>= 1) { if (tid < s) red[tid] += red[tid + s]; __syncthreads(); }
  float ls = __logf(red[0]);
  for (int c = tid; c < C; c += 256) out[(long long)row * C + c] = x[c] - m - ls;
}

// -------------------- host-side plumbing --------------------

static void* wsa(char* base, size_t& off, size_t bytes) {
  void* p = base + off;
  off = (off + bytes + 255) & ~(size_t)255;
  return p;
}

static int ilog2i(int v) { int s = 0; while ((1 << s) < v) ++s; return s; }

static void cvt(const float* in, bf16_t* out, long long n, hipStream_t s) {
  long long b = (n + 255) / 256; if (b > 1048576) b = 1048576;
  k_f32_to_bf16<<<dim3((unsigned)b), 256, 0, s>>>(in, out, n);
}
static void zero(float* p, long long n, hipStream_t s) {
  long long b = (n + 255) / 256; if (b > 1048576) b = 1048576;
  k_zero_f32<<<dim3((unsigned)b), 256, 0, s>>>(p, n);
}
static void scatter(const bf16_t* f, const int* idx, float* acc, float* cnt, int rows, int ncol,
                    hipStream_t s) {
  long long n = (long long)rows * ncol;
  long long b = (n + 255) / 256; if (b > 1048576) b = 1048576;
  k_scatter_add<<<dim3((unsigned)b), 256, 0, s>>>(f, idx, acc, cnt, rows, ilog2i(ncol));
}
static void meanb(const float* acc, const float* cnt, bf16_t* out, int rows, int ncol,
                  hipStream_t s) {
  long long n = (long long)rows * ncol;
  long long b = (n + 255) / 256; if (b > 1048576) b = 1048576;
  k_mean_to_bf16<<<dim3((unsigned)b), 256, 0, s>>>(acc, cnt, out, rows, ilog2i(ncol));
}
static void gemm(const ConcatA& A, const bf16_t* W, const float* bias, void* C,
                 int M, int N, int K, int relu, int outbf16, hipStream_t s) {
  dim3 g((unsigned)((N + 127) / 128), (unsigned)((M + 127) / 128));
  if (relu && outbf16)      k_gemm_wmma<1, 1><<<g, 256, 0, s>>>(A, W, bias, C, M, N, K);
  else if (relu)            k_gemm_wmma<1, 0><<<g, 256, 0, s>>>(A, W, bias, C, M, N, K);
  else                      k_gemm_wmma<0, 0><<<g, 256, 0, s>>>(A, W, bias, C, M, N, K);
}

struct LayerP { const float *We, *be, *Wn, *bn, *Wg, *bg; };

// (node_in, edge_in, edge_out, node_out, global_in_u, global_out)
static const int CFGh[4][6] = {
  { 39,  13, 512, 128,  16, 128},
  {128, 512, 512, 128, 128, 128},
  {128, 512, 512, 128, 128, 128},
  {128, 512, 128, 128, 128, 256},
};

extern "C" void kernel_launch(void* const* d_in, const int* in_sizes, int n_in,
                              void* d_out, int out_size, void* d_ws, size_t ws_size,
                              hipStream_t stream) {
  (void)in_sizes; (void)n_in; (void)out_size; (void)ws_size;

  const float* rx  = (const float*)d_in[0];
  const float* reE = (const float*)d_in[1];
  const int*   rc  = (const int*)d_in[2];
  const int*   rb  = (const int*)d_in[3];
  const float* rg  = (const float*)d_in[4];
  const float* px  = (const float*)d_in[5];
  const float* peE = (const float*)d_in[6];
  const int*   pc  = (const int*)d_in[7];
  const int*   pb  = (const int*)d_in[8];
  const float* pg  = (const float*)d_in[9];

  int pi = 10;
  LayerP LR[4], LP[4];
  for (int i = 0; i < 4; ++i) {
    LR[i].We = (const float*)d_in[pi++]; LR[i].be = (const float*)d_in[pi++];
    LR[i].Wn = (const float*)d_in[pi++]; LR[i].bn = (const float*)d_in[pi++];
    LR[i].Wg = (const float*)d_in[pi++]; LR[i].bg = (const float*)d_in[pi++];
  }
  for (int i = 0; i < 4; ++i) {
    LP[i].We = (const float*)d_in[pi++]; LP[i].be = (const float*)d_in[pi++];
    LP[i].Wn = (const float*)d_in[pi++]; LP[i].bn = (const float*)d_in[pi++];
    LP[i].Wg = (const float*)d_in[pi++]; LP[i].bg = (const float*)d_in[pi++];
  }
  const float* W1 = (const float*)d_in[pi++];
  const float* b1 = (const float*)d_in[pi++];
  const float* W2 = (const float*)d_in[pi++];
  const float* b2 = (const float*)d_in[pi++];

  // ---- workspace layout ----
  char* base = (char*)d_ws; size_t off = 0;
  bf16_t* xb[2]; bf16_t* eb[2]; bf16_t* ubr[2]; bf16_t* ubp[2];
  xb[0]  = (bf16_t*)wsa(base, off, (size_t)N_NODES * 128 * sizeof(bf16_t));
  xb[1]  = (bf16_t*)wsa(base, off, (size_t)N_NODES * 128 * sizeof(bf16_t));
  eb[0]  = (bf16_t*)wsa(base, off, (size_t)N_EDGES * 512 * sizeof(bf16_t));
  eb[1]  = (bf16_t*)wsa(base, off, (size_t)N_EDGES * 512 * sizeof(bf16_t));
  float*  agg   = (float*) wsa(base, off, (size_t)N_NODES * 512 * sizeof(float));
  bf16_t* aggb  = (bf16_t*)wsa(base, off, (size_t)N_NODES * 512 * sizeof(bf16_t));
  float*  cnt   = (float*) wsa(base, off, (size_t)N_NODES * sizeof(float));
  float*  pool  = (float*) wsa(base, off, (size_t)N_GRAPHS * 256 * sizeof(float));
  bf16_t* poolb = (bf16_t*)wsa(base, off, (size_t)N_GRAPHS * 256 * sizeof(bf16_t));
  float*  cntg  = (float*) wsa(base, off, (size_t)N_GRAPHS * sizeof(float));
  ubr[0] = (bf16_t*)wsa(base, off, (size_t)N_GRAPHS * 256 * sizeof(bf16_t));
  ubr[1] = (bf16_t*)wsa(base, off, (size_t)N_GRAPHS * 256 * sizeof(bf16_t));
  ubp[0] = (bf16_t*)wsa(base, off, (size_t)N_GRAPHS * 256 * sizeof(bf16_t));
  ubp[1] = (bf16_t*)wsa(base, off, (size_t)N_GRAPHS * 256 * sizeof(bf16_t));
  bf16_t* wWb    = (bf16_t*)wsa(base, off, (size_t)768 * 512 * sizeof(bf16_t));
  bf16_t* hbuf   = (bf16_t*)wsa(base, off, (size_t)N_GRAPHS * 512 * sizeof(bf16_t));
  float*  logits = (float*) wsa(base, off, (size_t)N_GRAPHS * OUTC * sizeof(float));

  auto run_stream = [&](const float* x0, const float* e0, const int* ci, const int* bidx,
                        const float* u0, const LayerP* L, bf16_t** ub) -> int {
    cvt(x0, xb[0], (long long)N_NODES * FN, stream);
    cvt(e0, eb[0], (long long)N_EDGES * FE, stream);
    cvt(u0, ub[0], (long long)N_GRAPHS * FG, stream);
    int xc = 0, ec = 0, uc = 0;
    int xd = FN, ed = FE, ud = FG;
    const int* rowI = ci;
    const int* colI = ci + N_EDGES;
    for (int l = 0; l < 4; ++l) {
      const int nx = CFGh[l][0], ein = CFGh[l][1], eout = CFGh[l][2],
                nout = CFGh[l][3], gu = CFGh[l][4], gout = CFGh[l][5];
      // edge model: relu(concat(x[row], x[col], e) @ We + be)
      const int Ke = 2 * nx + ein;
      cvt(L[l].We, wWb, (long long)Ke * eout, stream);
      ConcatA Ae = { xb[xc], rowI, xd, nx,  xb[xc], colI, xd, 2 * nx,  eb[ec], ed };
      gemm(Ae, wWb, L[l].be, eb[ec ^ 1], N_EDGES, eout, Ke, 1, 1, stream);
      ec ^= 1; ed = eout;
      // node model: agg = seg_mean(e_new, col); relu(concat(x, agg) @ Wn + bn)
      zero(agg, (long long)N_NODES * eout, stream);
      zero(cnt, N_NODES, stream);
      scatter(eb[ec], colI, agg, cnt, N_EDGES, eout, stream);
      meanb(agg, cnt, aggb, N_NODES, eout, stream);
      const int Kn = nx + eout;
      cvt(L[l].Wn, wWb, (long long)Kn * nout, stream);
      ConcatA An = { xb[xc], nullptr, xd, nx,  aggb, nullptr, eout, Kn,  nullptr, 0 };
      gemm(An, wWb, L[l].bn, xb[xc ^ 1], N_NODES, nout, Kn, 1, 1, stream);
      xc ^= 1; xd = nout;
      // global model: pool = seg_mean(x_new, b); relu(concat(u, pool) @ Wg + bg)
      zero(pool, (long long)N_GRAPHS * nout, stream);
      zero(cntg, N_GRAPHS, stream);
      scatter(xb[xc], bidx, pool, cntg, N_NODES, nout, stream);
      meanb(pool, cntg, poolb, N_GRAPHS, nout, stream);
      const int Kg = gu + nout;
      cvt(L[l].Wg, wWb, (long long)Kg * gout, stream);
      ConcatA Ag = { ub[uc], nullptr, ud, gu,  poolb, nullptr, nout, Kg,  nullptr, 0 };
      gemm(Ag, wWb, L[l].bg, ub[uc ^ 1], N_GRAPHS, gout, Kg, 1, 1, stream);
      uc ^= 1; ud = gout;
    }
    return uc;
  };

  int ur = run_stream(rx, reE, rc, rb, rg, LR, ubr);
  int up = run_stream(px, peE, pc, pb, pg, LP, ubp);

  // readout: y = relu(concat(rg, pg) @ W1 + b1); out = log_softmax(y @ W2 + b2)
  cvt(W1, wWb, 512LL * 512, stream);
  ConcatA Af = { ubr[ur], nullptr, 256, 256,  ubp[up], nullptr, 256, 512,  nullptr, 0 };
  gemm(Af, wWb, b1, hbuf, N_GRAPHS, 512, 512, 1, 1, stream);

  cvt(W2, wWb, 512LL * OUTC, stream);
  ConcatA A2 = { hbuf, nullptr, 512, 512,  nullptr, nullptr, 0, 512,  nullptr, 0 };
  gemm(A2, wWb, b2, logits, N_GRAPHS, OUTC, 512, 0, 0, stream);

  k_log_softmax<<<dim3(N_GRAPHS), 256, 0, stream>>>(logits, (float*)d_out, OUTC);
}
